// DownsampleModule_18915035971605
// MI455X (gfx1250) — compile-verified, matrix operand order checked by reading, not run on hardware
//
#include <hip/hip_runtime.h>

typedef __attribute__((ext_vector_type(16))) _Float16 v16h;
typedef __attribute__((ext_vector_type(8)))  float    v8f;

#define N_NODES  100000
#define N_EDGES  1600000
#define C_IN     32
#define C_OUT    64
#define KDIM     128          // 4*C_IN
#define PROPC    96           // 3*C_IN
#define VOXD     16
#define NCLUST   1024         // B * VOXD * VOXD
#define VOX_INV  16.0f        // 1 / 0.0625
#define ROWS_PB  32           // rows (nodes) per block in the GEMM kernel

// ---------------------------------------------------------------------------
// Zero-fill (accumulators + pooled output regions must be zeroed every launch)
// ---------------------------------------------------------------------------
__global__ void zero_kernel(float* __restrict__ p, long n) {
    long i = (long)blockIdx.x * blockDim.x + threadIdx.x;
    long stride = (long)gridDim.x * blockDim.x;
    for (; i < n; i += stride) p[i] = 0.0f;
}

// ---------------------------------------------------------------------------
// One-time W (f32 [64,128]) -> Wh (f16) conversion, amortized across all
// GEMM blocks (removes 8K VALU converts + 32KB L2 reads per block).
// ---------------------------------------------------------------------------
__global__ void convert_w_kernel(const float* __restrict__ W,
                                 _Float16* __restrict__ Wh) {
    int i = blockIdx.x * blockDim.x + threadIdx.x;
    if (i < C_OUT * KDIM) Wh[i] = (_Float16)W[i];
}

// ---------------------------------------------------------------------------
// Edge scatter: one wave32 per edge, lane = channel. Coalesced 128B row
// gathers of x (resident in 192MB L2), 3 float atomics per lane + 1 count.
// This phase dominates the roofline (atomic/L2-latency bound).
// ---------------------------------------------------------------------------
__global__ __launch_bounds__(256)
void edge_scatter_kernel(const float* __restrict__ x,
                         const int*   __restrict__ ei,
                         const float* __restrict__ pos,
                         float* __restrict__ sum,
                         float* __restrict__ cnt) {
    long gid = (long)blockIdx.x * blockDim.x + threadIdx.x;
    long e   = gid >> 5;             // one wave per edge
    int  c   = threadIdx.x & 31;     // lane = channel
    if (e >= N_EDGES) return;

    int row = ei[e];                 // j (source)
    int col = ei[N_EDGES + e];       // i (target)

    // prefetch a later edge's source row (gfx1250 global_prefetch_b8)
    if (e + 4 < N_EDGES) {
        int nrow = ei[e + 4];
        __builtin_prefetch(x + (long)nrow * C_IN, 0, 1);
    }

    float dx = pos[2 * col]     - pos[2 * row];
    float dy = pos[2 * col + 1] - pos[2 * row + 1];
    float inv = 1.0f / (dx * dx + dy * dy + 0.01f);   // R_CORR == 1
    float wx = dx * inv, wy = dy * inv;

    float xj = x[(long)row * C_IN + c];
    float xi = x[(long)col * C_IN + c];
    float vd = xi - xj;

    float* s = sum + (long)col * PROPC;
    atomicAdd(s + c,            vd * wx);
    atomicAdd(s + C_IN + c,     vd * wy);
    atomicAdd(s + 2 * C_IN + c, xj);
    if (c == 0) atomicAdd(cnt + col, 1.0f);
}

// ---------------------------------------------------------------------------
// Per-node voxel-pos accumulation + passthrough outputs (pos, batch)
// ---------------------------------------------------------------------------
__global__ __launch_bounds__(256)
void node_pos_kernel(const float* __restrict__ pos,
                     const int*   __restrict__ batch,
                     float* __restrict__ possum,
                     float* __restrict__ ccnt,
                     float* __restrict__ out_pos,
                     float* __restrict__ out_batch) {
    long i = (long)blockIdx.x * blockDim.x + threadIdx.x;
    long stride = (long)gridDim.x * blockDim.x;
    for (; i < N_NODES; i += stride) {
        float px = pos[2 * i], py = pos[2 * i + 1];
        int vx = min(max((int)floorf(px * VOX_INV), 0), VOXD - 1);
        int vy = min(max((int)floorf(py * VOX_INV), 0), VOXD - 1);
        int cl = batch[i] * (VOXD * VOXD) + vy * VOXD + vx;
        atomicAdd(possum + 2 * cl,     px);
        atomicAdd(possum + 2 * cl + 1, py);
        atomicAdd(ccnt + cl, 1.0f);
        out_pos[2 * i]     = px;
        out_pos[2 * i + 1] = py;
        out_batch[i] = (float)batch[i];
    }
}

// ---------------------------------------------------------------------------
// Fused: prop = [sum/max(cnt,1) | x]  (32 rows/block, f16 in LDS)
//        h    = relu(prop @ W^T + b)  via v_wmma_f32_16x16x32_f16
//        pooled_x = segment_max(h)    via uint atomicMax (h >= 0)
// 256 threads = 8 waves: wave (rt,ct) owns row-tile rt (16 rows) and
// column-tile ct (16 cols). W tile pulled in via GLOBAL_LOAD_ASYNC_TO_LDS
// (ASYNCcnt path, no VGPR staging), fenced with s_wait_asynccnt.
// ---------------------------------------------------------------------------
__global__ __launch_bounds__(256)
void gemm_pool_kernel(const float* __restrict__ x,
                      const float* __restrict__ sum,
                      const float* __restrict__ cnt,
                      const _Float16* __restrict__ Wh,
                      const float* __restrict__ bvec,
                      const float* __restrict__ pos,
                      const int*   __restrict__ batch,
                      float* __restrict__ pooled) {
    __shared__ _Float16 As[ROWS_PB * KDIM];   // 8 KB
    __shared__ _Float16 Ws[C_OUT * KDIM];     // 16 KB
    __shared__ int      cls[ROWS_PB];

    int  tid      = threadIdx.x;
    long nodeBase = (long)blockIdx.x * ROWS_PB;

    // --- async copy of the f16 W tile into LDS (1024 x b128 = 16 KB) ------
    {
        unsigned long long gbase = (unsigned long long)(uintptr_t)Wh;
        // low 32 bits of a generic LDS pointer are the LDS byte offset
        unsigned lbase = (unsigned)(uintptr_t)(&Ws[0]);
#pragma unroll
        for (int s = 0; s < 4; ++s) {
            int byteOff = (tid + s * 256) * 16;
            unsigned long long ga = gbase + byteOff;
            unsigned la = lbase + byteOff;
            asm volatile("global_load_async_to_lds_b128 %0, %1, off"
                         :: "v"(la), "v"(ga) : "memory");
        }
    }

    // --- A tile: prop[r][k] = k<96 ? sum[node][k]/max(cnt,1) : x[node][k-96]
    for (int i = tid; i < ROWS_PB * KDIM; i += 256) {
        int r = i >> 7;
        int k = i & (KDIM - 1);
        long node = nodeBase + r;
        float v = 0.0f;
        if (node < N_NODES) {
            if (k < PROPC) {
                float cinv = 1.0f / fmaxf(cnt[node], 1.0f);
                v = sum[node * PROPC + k] * cinv;
            } else {
                v = x[node * C_IN + (k - PROPC)];
            }
        }
        As[i] = (_Float16)v;
    }

    // cluster id per row
    if (tid < ROWS_PB) {
        long node = nodeBase + tid;
        int cl = -1;
        if (node < N_NODES) {
            float px = pos[2 * node], py = pos[2 * node + 1];
            int vx = min(max((int)floorf(px * VOX_INV), 0), VOXD - 1);
            int vy = min(max((int)floorf(py * VOX_INV), 0), VOXD - 1);
            cl = batch[node] * (VOXD * VOXD) + vy * VOXD + vx;
        }
        cls[tid] = cl;
    }

    asm volatile("s_wait_asynccnt 0" ::: "memory");   // W tile landed in LDS
    __syncthreads();

    int wave  = tid >> 5;
    int lane  = tid & 31;
    int rt    = wave >> 2;           // row-tile (0..1)
    int cn    = (wave & 3) * 16;     // column-tile base (0..48)
    int nloc  = lane & 15;
    int kbase = (lane >> 4) * 8;     // ISA 16-bit A/B fragment K grouping

    // C init = bias broadcast (bias depends only on column)
    float bv = bvec[cn + nloc];
    v8f acc;
#pragma unroll
    for (int r = 0; r < 8; ++r) acc[r] = bv;

    // K = 128 as 4 x (16x16x32) WMMA
#pragma unroll
    for (int kc = 0; kc < 4; ++kc) {
        const _Float16* arow = As + (rt * 16 + (lane & 15)) * KDIM + kc * 32;
        const _Float16* brow = Ws + (cn + nloc) * KDIM + kc * 32;
        v16h a, bb;
#pragma unroll
        for (int e2 = 0; e2 < 16; ++e2) {
            int k = (e2 < 8) ? (kbase + e2) : (16 + kbase + (e2 - 8));
            a[e2]  = arow[k];
            bb[e2] = brow[k];
        }
        acc = __builtin_amdgcn_wmma_f32_16x16x32_f16(
            /*neg_a=*/false, a, /*neg_b=*/false, bb,
            /*c_mod=*/(short)0, acc, /*reuse_a=*/false, /*reuse_b=*/false);
    }

    // relu + voxel max-pool (h >= 0, so uint compare == float compare)
#pragma unroll
    for (int r = 0; r < 8; ++r) {
        int mrow = r + ((lane >> 4) * 8);    // C/D layout: row = vgpr + 8*(lane/16)
        int cl = cls[rt * 16 + mrow];
        if (cl >= 0) {
            float v = fmaxf(acc[r], 0.0f);
            atomicMax((unsigned int*)(pooled + (long)cl * C_OUT + cn + nloc),
                      __float_as_uint(v));
        }
    }
}

// ---------------------------------------------------------------------------
// pooled_pos = possum / max(ccnt, 1)
// ---------------------------------------------------------------------------
__global__ void pos_finalize_kernel(const float* __restrict__ possum,
                                    const float* __restrict__ ccnt,
                                    float* __restrict__ out_pp) {
    int i = blockIdx.x * blockDim.x + threadIdx.x;   // over NCLUST*2
    if (i >= NCLUST * 2) return;
    int cl = i >> 1;
    out_pp[i] = possum[i] / fmaxf(ccnt[cl], 1.0f);
}

// ---------------------------------------------------------------------------
extern "C" void kernel_launch(void* const* d_in, const int* in_sizes, int n_in,
                              void* d_out, int out_size, void* d_ws, size_t ws_size,
                              hipStream_t stream) {
    const float* x     = (const float*)d_in[0];   // [N, 32]
    const int*   ei    = (const int*)  d_in[1];   // [2, E]
    const float* pos   = (const float*)d_in[2];   // [N, 2]
    const int*   batch = (const int*)  d_in[3];   // [N]
    const float* W     = (const float*)d_in[4];   // [64, 128]
    const float* bvec  = (const float*)d_in[5];   // [64]

    float* out        = (float*)d_out;
    float* out_pooled = out;                                   // 1024*64
    float* out_pp     = out + NCLUST * C_OUT;                  // 1024*2
    float* out_pos    = out_pp + NCLUST * 2;                   // N*2
    float* out_batch  = out_pos + (long)N_NODES * 2;           // N

    float* sum    = (float*)d_ws;                              // N*96
    float* cnt    = sum + (long)N_NODES * PROPC;               // N
    float* possum = cnt + N_NODES;                             // 1024*2
    float* ccnt   = possum + NCLUST * 2;                       // 1024
    _Float16* wh  = (_Float16*)(ccnt + NCLUST);                // 64*128 f16

    // 1) zero accumulators + pooled output head (deterministic re-launch)
    long zws = (long)N_NODES * PROPC + N_NODES + NCLUST * 3;
    zero_kernel<<<2048, 256, 0, stream>>>((float*)d_ws, zws);
    zero_kernel<<<256, 256, 0, stream>>>(out, NCLUST * C_OUT + NCLUST * 2);

    // 2) W -> f16 (once)
    convert_w_kernel<<<(C_OUT * KDIM + 255) / 256, 256, 0, stream>>>(W, wh);

    // 3) edge scatter-sum: one wave per edge
    {
        long threads = (long)N_EDGES * 32;
        int blocks = (int)((threads + 255) / 256);
        edge_scatter_kernel<<<blocks, 256, 0, stream>>>(x, ei, pos, sum, cnt);
    }

    // 4) voxel pos accumulation + passthrough outputs
    node_pos_kernel<<<1024, 256, 0, stream>>>(pos, batch, possum, ccnt,
                                              out_pos, out_batch);

    // 5) fused WMMA GEMM + relu + voxel max-pool
    {
        int blocks = (N_NODES + ROWS_PB - 1) / ROWS_PB;
        gemm_pool_kernel<<<blocks, 256, 0, stream>>>(x, sum, cnt, wh, bvec,
                                                     pos, batch, out_pooled);
    }

    // 6) pooled_pos finalize
    pos_finalize_kernel<<<(NCLUST * 2 + 255) / 256, 256, 0, stream>>>(
        possum, ccnt, out_pp);
}
